// MatchNet_21689584845320
// MI455X (gfx1250) — matrix operand order repaired and checked by reading, not compiled, not run on hardware
//
#include <hip/hip_runtime.h>
#include <hip/hip_bf16.h>

// ---------------------------------------------------------------------------
// MatchNet kNN-softmax on MI455X (gfx1250, wave32, WMMA bf16)
// ---------------------------------------------------------------------------
#define B_Q   1024
#define N_C   100000
#define DIN   256
#define DIM   512
#define K_NN  32
#define NT16  (N_C / 16)   // 6250 candidate column tiles
#define S_SL  64           // N-slices per query block
#define FBIG  3.402823466e+38f

typedef __bf16 v16bf __attribute__((ext_vector_type(16)));
typedef __bf16 v8bf  __attribute__((ext_vector_type(8)));
typedef float  v8f   __attribute__((ext_vector_type(8)));

// Wave-wide insert of (d, y) into a 32-entry list distributed one slot/lane.
// thr tracks the current max of the list (uniform across lanes).
__device__ __forceinline__ void wave_insert(float d, float y,
                                            float& slotD, float& slotY,
                                            float& thr, int lane) {
  float mx = slotD; int who = lane;
  #pragma unroll
  for (int off = 16; off; off >>= 1) {
    const float ov = __shfl_xor(mx, off);
    const int   ow = __shfl_xor(who, off);
    if (ov > mx || (ov == mx && ow < who)) { mx = ov; who = ow; }
  }
  if (d < mx) {                       // uniform branch
    if (lane == who) { slotD = d; slotY = y; }
    float nm = slotD;
    #pragma unroll
    for (int off = 16; off; off >>= 1) nm = fmaxf(nm, __shfl_xor(nm, off));
    thr = nm;
  }
}

// ---------------------------------------------------------------------------
// Kernel 1/2: encoder  E[r, :] = X[r, :] @ W.T + b  (bf16 out) + f32 row norms
// grid.x = R/16 row tiles, 256 threads = 8 waves, wave w owns columns w*64..+64
// ---------------------------------------------------------------------------
__global__ __launch_bounds__(256) void encode_k(const float* __restrict__ X,
                                                const float* __restrict__ W,
                                                const float* __restrict__ bias,
                                                __bf16* __restrict__ E,
                                                float* __restrict__ norms) {
  __shared__ float snorm[16];
  const int tid  = threadIdx.x;
  const int lane = tid & 31;
  const int wv   = tid >> 5;      // 0..7
  const int h    = lane >> 4;     // lane half (A/B layout split)
  const int lm   = lane & 15;
  const long r0  = (long)blockIdx.x * 16;
  const int  nb  = wv * 64;

  if (tid < 16) snorm[tid] = 0.0f;
  __syncthreads();

  v8f acc[4];
  #pragma unroll
  for (int t = 0; t < 4; ++t)
    #pragma unroll
    for (int i = 0; i < 8; ++i) acc[t][i] = 0.0f;

  const float* arow = X + (r0 + lm) * DIN;
  #pragma unroll
  for (int kc = 0; kc < DIN / 32; ++kc) {
    const int kk = kc * 32;
    // A: 16x32 bf16 tile; lane half h supplies K = kk+h*8..+8 and kk+16+h*8..+8
    v8f alo = *(const v8f*)(arow + kk + h * 8);
    v8f ahi = *(const v8f*)(arow + kk + 16 + h * 8);
    v16bf a;
    #pragma unroll
    for (int i = 0; i < 8; ++i) { a[i] = (__bf16)alo[i]; a[8 + i] = (__bf16)ahi[i]; }
    #pragma unroll
    for (int t = 0; t < 4; ++t) {
      // B[k][n] = W[n][k]: K = kk+h*8..+8 and kk+16+h*8..+8 of row n
      const int n = nb + t * 16 + lm;
      const float* wrow = W + (long)n * DIN + kk + h * 8;
      v8f blo = *(const v8f*)(wrow);
      v8f bhi = *(const v8f*)(wrow + 16);
      v16bf bm;
      #pragma unroll
      for (int i = 0; i < 8; ++i) { bm[i] = (__bf16)blo[i]; bm[8 + i] = (__bf16)bhi[i]; }
      acc[t] = __builtin_amdgcn_wmma_f32_16x16x32_bf16(false, a, false, bm,
                                                       (short)0, acc[t], false, false);
    }
  }

  // bias + store bf16; keep biased values for the norm reduction
  #pragma unroll
  for (int t = 0; t < 4; ++t) {
    const int n = nb + t * 16 + lm;
    const float bn = bias[n];
    #pragma unroll
    for (int j = 0; j < 8; ++j) {
      const float v = acc[t][j] + bn;
      const int m = h * 8 + j;           // C layout: vgpr j, lane half -> row
      E[(r0 + m) * DIM + n] = (__bf16)v;
      acc[t][j] = v;
    }
  }
  // row norms: reduce v^2 over this wave's 64 columns, then LDS-atomic across waves
  #pragma unroll
  for (int j = 0; j < 8; ++j) {
    float s = 0.0f;
    #pragma unroll
    for (int t = 0; t < 4; ++t) s += acc[t][j] * acc[t][j];
    s += __shfl_xor(s, 1); s += __shfl_xor(s, 2);
    s += __shfl_xor(s, 4); s += __shfl_xor(s, 8);
    if (lm == 0) atomicAdd(&snorm[h * 8 + j], s);
  }
  __syncthreads();
  if (tid < 16) norms[r0 + tid] = snorm[tid];
}

// ---------------------------------------------------------------------------
// Kernel 3: fused distance GEMM + streaming per-query top-32
// 64 query-blocks x 64 N-slices; 8 waves/block (all 8 share one query block)
// ---------------------------------------------------------------------------
__global__ __launch_bounds__(256) void knn_k(const __bf16* __restrict__ XE,
                                             const float* __restrict__ XN,
                                             const __bf16* __restrict__ CE,
                                             const float* __restrict__ CN,
                                             const float* __restrict__ CY,
                                             float2* __restrict__ parts) {
  __shared__ __align__(16) __bf16 sA[16 * 520];  // 16 rows, pad 8 -> stride 520
  const int tid   = threadIdx.x;
  const int lane  = tid & 31;
  const int wv    = tid >> 5;
  const int h     = lane >> 4;
  const int lm    = lane & 15;
  const int wid   = blockIdx.x * 8 + wv;
  const int qb    = wid / S_SL;   // uniform within block (S_SL multiple of 8)
  const int slice = wid % S_SL;

  {  // cooperative stage of the 16 query rows (16x512 bf16) into LDS
    const int row = tid >> 4, seg = tid & 15;
    const uint4* src = (const uint4*)(XE + ((long)(qb * 16 + row)) * DIM + seg * 32);
    uint4* dst = (uint4*)(sA + row * 520 + seg * 32);
    #pragma unroll
    for (int i = 0; i < 4; ++i) dst[i] = src[i];
  }
  __syncthreads();

  float xn[8];
  #pragma unroll
  for (int j = 0; j < 8; ++j) xn[j] = XN[qb * 16 + h * 8 + j];

  float ld[16], ly[16], thr[16];
  #pragma unroll
  for (int m = 0; m < 16; ++m) { ld[m] = FBIG; ly[m] = 0.0f; thr[m] = FBIG; }

  const __bf16* srow = sA + lm * 520;   // this lane's A row (M = lane & 15)

  for (int t = slice; t < NT16; t += S_SL) {
    const long ncol = (long)t * 16 + lm;
    const float cn = CN[ncol];
    const float yv = CY[ncol];
    const __bf16* crow = CE + ncol * DIM;
    if (t + S_SL < NT16)                       // global_prefetch_b8 next tile
      __builtin_prefetch(CE + (ncol + (long)S_SL * 16) * DIM, 0, 1);

    v8f acc;
    #pragma unroll
    for (int i = 0; i < 8; ++i) acc[i] = 0.0f;
    #pragma unroll
    for (int kc = 0; kc < DIM / 32; ++kc) {
      const int kk = kc * 32;
      v8bf alo = *(const v8bf*)(srow + kk + h * 8);
      v8bf ahi = *(const v8bf*)(srow + kk + 16 + h * 8);
      v16bf a;
      #pragma unroll
      for (int i = 0; i < 8; ++i) { a[i] = alo[i]; a[8 + i] = ahi[i]; }
      v8bf blo = *(const v8bf*)(crow + kk + h * 8);  // K = kk+h*8..+8 and kk+16+h*8..+8 of col n
      v8bf bhi = *(const v8bf*)(crow + kk + 16 + h * 8);
      v16bf bm;
      #pragma unroll
      for (int i = 0; i < 8; ++i) { bm[i] = blo[i]; bm[8 + i] = bhi[i]; }
      acc = __builtin_amdgcn_wmma_f32_16x16x32_bf16(false, a, false, bm,
                                                    (short)0, acc, false, false);
    }
    // epilogue: dist = sqrt(max(||x||^2 + ||c||^2 - 2 x.c, 0)); threshold+insert
    #pragma unroll
    for (int j = 0; j < 8; ++j) {
      const float d2 = xn[j] + cn - 2.0f * acc[j];
      const float d  = sqrtf(fmaxf(d2, 0.0f));
      const float myThr = h ? thr[j + 8] : thr[j];
      unsigned long long bal = __ballot(d < myThr);
      unsigned low  = (unsigned)(bal & 0xFFFFull);          // queries m = j
      unsigned high = (unsigned)((bal >> 16) & 0xFFFFull);  // queries m = j+8
      while (low) {
        const int s = __ffs(low) - 1; low &= low - 1;
        wave_insert(__shfl(d, s), __shfl(yv, s), ld[j], ly[j], thr[j], lane);
      }
      while (high) {
        const int s = (__ffs(high) - 1) + 16; high &= high - 1;
        wave_insert(__shfl(d, s), __shfl(yv, s), ld[j + 8], ly[j + 8], thr[j + 8], lane);
      }
    }
  }
  #pragma unroll
  for (int m = 0; m < 16; ++m)
    parts[(((long)(qb * 16 + m)) * S_SL + slice) * 32 + lane] =
        make_float2(ld[m], ly[m]);
}

// ---------------------------------------------------------------------------
// Kernel 4: merge 64 partial top-32 lists per query, softmax(-dist) * y
// one wave per query
// ---------------------------------------------------------------------------
__global__ __launch_bounds__(256) void merge_k(const float2* __restrict__ parts,
                                               float* __restrict__ out) {
  const int lane = threadIdx.x & 31;
  const int q = blockIdx.x * 8 + (threadIdx.x >> 5);
  float bd = FBIG, by = 0.0f, thr = FBIG;
  const float2* src = parts + (long)q * (S_SL * 32);
  for (int e = lane; e < S_SL * 32; e += 32) {
    const float2 p = src[e];
    unsigned mask = (unsigned)(__ballot(p.x < thr) & 0xFFFFFFFFull);
    while (mask) {
      const int s = __ffs(mask) - 1; mask &= mask - 1;
      wave_insert(__shfl(p.x, s), __shfl(p.y, s), bd, by, thr, lane);
    }
  }
  // softmax over -dist of the 32 survivors (one per lane)
  const float neg = -bd;
  float mx = neg;
  #pragma unroll
  for (int off = 16; off; off >>= 1) mx = fmaxf(mx, __shfl_xor(mx, off));
  const float w = __expf(neg - mx);
  float sw = w, swy = w * by;
  #pragma unroll
  for (int off = 16; off; off >>= 1) {
    sw  += __shfl_xor(sw, off);
    swy += __shfl_xor(swy, off);
  }
  if (lane == 0) out[q] = swy / sw;
}

// ---------------------------------------------------------------------------
extern "C" void kernel_launch(void* const* d_in, const int* in_sizes, int n_in,
                              void* d_out, int out_size, void* d_ws, size_t ws_size,
                              hipStream_t stream) {
  (void)in_sizes; (void)n_in; (void)out_size; (void)ws_size;
  const float* x    = (const float*)d_in[0];  // [1024, 256]
  const float* cx   = (const float*)d_in[1];  // [100000, 256]
  const float* cy   = (const float*)d_in[2];  // [100000]
  const float* W    = (const float*)d_in[3];  // [512, 256]
  const float* bias = (const float*)d_in[4];  // [512]
  float* out        = (float*)d_out;          // [1024]

  char* ws = (char*)d_ws;
  size_t off = 0;
  auto alloc = [&](size_t bytes) -> void* {
    off = (off + 255) & ~(size_t)255;
    void* p = ws + off; off += bytes; return p;
  };
  __bf16* ce   = (__bf16*)alloc((size_t)N_C * DIM * sizeof(__bf16)); // 102.4 MB (L2-resident)
  float*  cn   = (float*) alloc((size_t)N_C * sizeof(float));
  __bf16* xe   = (__bf16*)alloc((size_t)B_Q * DIM * sizeof(__bf16));
  float*  xnrm = (float*) alloc((size_t)B_Q * sizeof(float));
  float2* prt  = (float2*)alloc((size_t)B_Q * S_SL * 32 * sizeof(float2)); // 16.8 MB

  encode_k<<<N_C / 16, 256, 0, stream>>>(cx, W, bias, ce, cn);
  encode_k<<<B_Q / 16, 256, 0, stream>>>(x,  W, bias, xe, xnrm);
  knn_k  <<<(B_Q / 16) * S_SL / 8, 256, 0, stream>>>(xe, xnrm, ce, cn, cy, prt);
  merge_k<<<B_Q / 8, 256, 0, stream>>>(prt, out);
}
